// Decoder_29454885716716
// MI455X (gfx1250) — matrix operand order repaired and checked by reading, not compiled, not run on hardware
//
#include <hip/hip_runtime.h>

// B=16, T=2048, S=2048, D=1024, H=24 (padded to HP=32)
#define Bn 16
#define Tn 2048
#define Sn 2048
#define Dn 1024
#define HP 32

// k_attn LDS staging geometry
#define CH   128                 // s-chunk staged per double-buffer slot
#define KROW 40                  // padded kh row stride in LDS (f16) -> bank-conflict free A reads
#define VROW 136                 // padded vt row stride in LDS (f16) -> lanes land on distinct banks
#define SVTE (2 * CH * KROW)     // element offset of vt staging area in smem
#define NCHUNK (Sn / CH)
#define LDS_BYTES ((2 * CH * KROW + 2 * HP * VROW) * 2)  // 37,888 B

typedef __attribute__((ext_vector_type(16))) _Float16 v16h;
typedef __attribute__((ext_vector_type(8)))  _Float16 v8h;
typedef __attribute__((ext_vector_type(8)))  float    v8f;

__device__ __forceinline__ float fast_exp2(float x) {
  return __builtin_amdgcn_exp2f(x);   // v_exp_f32 (base-2 hardware exp)
}

__device__ __forceinline__ v8f wmma16(v16h a, v16h b, v8f c) {
  return __builtin_amdgcn_wmma_f32_16x16x32_f16(false, a, false, b, (short)0, c,
                                                false, false);
}

// Async global->LDS 16B copy (CDNA5, ASYNCcnt-tracked). lds = LDS byte address.
__device__ __forceinline__ void async_cp16(unsigned lds, const void* gp) {
  asm volatile("global_load_async_to_lds_b128 %0, %1, off"
               :: "v"(lds), "v"((unsigned long long)(uintptr_t)gp)
               : "memory");
}

// A-matrix (16x32 f16) load from row-major f16 [row][ld] (global or LDS).
// Lane L: row = row0 + (L&15); elems 0..7 -> K=8*hi+e ; 8..15 -> K=16+8*hi+e
__device__ __forceinline__ v16h load_a16(const _Float16* base, int ld, int row0, int k0) {
  int L = threadIdx.x & 31, m = L & 15, hi = L >> 4;
  const _Float16* p = base + (size_t)(row0 + m) * ld + k0 + 8 * hi;
  v8h lo = *(const v8h*)p;
  v8h hh = *(const v8h*)(p + 16);
  v16h a;
#pragma unroll
  for (int e = 0; e < 8; e++) { a[e] = lo[e]; a[e + 8] = hh[e]; }
  return a;
}

// Same A-tile, converting from f32 source on the fly.
__device__ __forceinline__ v16h load_a16_f32(const float* base, int ld, int row0, int k0) {
  int L = threadIdx.x & 31, m = L & 15, hi = L >> 4;
  const float* p = base + (size_t)(row0 + m) * ld + k0 + 8 * hi;
  v16h a;
#pragma unroll
  for (int e = 0; e < 8; e++) {
    a[e]     = (_Float16)p[e];
    a[e + 8] = (_Float16)p[e + 16];
  }
  return a;
}

// B-matrix (32x16 f16) from K-contiguous storage BT[n][ld]; two 16B loads per lane (K=8*hi+e, 16+8*hi+e).
__device__ __forceinline__ v16h load_b16(const _Float16* baseT, int ld, int n0, int k0) {
  int L = threadIdx.x & 31, n = L & 15, hi = L >> 4;
  const _Float16* p = baseT + (size_t)(n0 + n) * ld + k0 + 8 * hi;
  v8h lo = *(const v8h*)p, hh = *(const v8h*)(p + 16);
  v16h r;
#pragma unroll
  for (int e = 0; e < 8; e++) { r[e] = lo[e]; r[e + 8] = hh[e]; }
  return r;
}

// 16-f16 load as two 16B-aligned halves (for 272B-strided LDS rows).
__device__ __forceinline__ v16h load16h_2(const _Float16* p) {
  v8h a = *(const v8h*)p, b = *(const v8h*)(p + 16);
  v16h r;
#pragma unroll
  for (int e = 0; e < 8; e++) { r[e] = a[e]; r[e + 8] = b[e]; }
  return r;
}

// ---------------- Kernel 0: convert + pad weights to f16, K-contiguous ----------------
// wq gets log2(e) folded in so both softmax passes run pure exp2 (saves 1 VALU mul/score).
__global__ void k_wconv(const float* __restrict__ wq, const float* __restrict__ wk,
                        const float* __restrict__ wv, const float* __restrict__ wh,
                        _Float16* wqt, _Float16* wkt, _Float16* wvt, _Float16* wht) {
  int i = blockIdx.x * blockDim.x + threadIdx.x; // 0 .. 32*1024-1
  int h = i >> 10, k = i & (Dn - 1);
  _Float16 zq = (_Float16)0.f, zk = zq, zv = zq, zh = zq;
  if (h < 24) {
    zq = (_Float16)(wq[(size_t)k * 24 + h] * 1.44269504088896f); // * log2(e)
    zk = (_Float16)wk[(size_t)k * 24 + h];
    zv = (_Float16)wv[(size_t)k * 24 + h];
    zh = (_Float16)wh[(size_t)h * Dn + k];
  }
  wqt[(size_t)h * Dn + k] = zq;
  wkt[(size_t)h * Dn + k] = zk;
  wvt[(size_t)h * Dn + k] = zv;
  wht[(size_t)k * HP + h] = zh;
}

// ---------------- Kernel 1: Q projection: qh[row][32] = f16(text @ wq') ---------------
__global__ void k_projq(const float* __restrict__ X, const _Float16* __restrict__ Wt,
                        _Float16* __restrict__ out) {
  int wid  = blockIdx.x * (blockDim.x >> 5) + (threadIdx.x >> 5);
  int row0 = wid * 16;
  int L = threadIdx.x & 31, n = L & 15, hi = L >> 4;
  v8f c0 = {}, c1 = {};
  for (int kk = 0; kk < Dn; kk += 32) {
    v16h a = load_a16_f32(X, Dn, row0, kk);
    c0 = wmma16(a, load_b16(Wt, Dn, 0, kk), c0);
    c1 = wmma16(a, load_b16(Wt, Dn, 16, kk), c1);
  }
#pragma unroll
  for (int r = 0; r < 8; r++) {
    int row = row0 + r + 8 * hi;
    out[(size_t)row * HP + n]      = (_Float16)c0[r];
    out[(size_t)row * HP + 16 + n] = (_Float16)c1[r];
  }
}

// ------------- Kernel 2: fused K,V projection (encoded read once) ---------------------
__global__ void k_projkv(const float* __restrict__ X, const _Float16* __restrict__ Wkt,
                         const _Float16* __restrict__ Wvt,
                         _Float16* __restrict__ kh, _Float16* __restrict__ vt) {
  int wid  = blockIdx.x * (blockDim.x >> 5) + (threadIdx.x >> 5);
  int row0 = wid * 16;
  int L = threadIdx.x & 31, n = L & 15, hi = L >> 4;
  v8f ck0 = {}, ck1 = {}, cv0 = {}, cv1 = {};
  for (int kk = 0; kk < Dn; kk += 32) {
    v16h a = load_a16_f32(X, Dn, row0, kk);
    ck0 = wmma16(a, load_b16(Wkt, Dn, 0, kk), ck0);
    ck1 = wmma16(a, load_b16(Wkt, Dn, 16, kk), ck1);
    cv0 = wmma16(a, load_b16(Wvt, Dn, 0, kk), cv0);
    cv1 = wmma16(a, load_b16(Wvt, Dn, 16, kk), cv1);
  }
#pragma unroll
  for (int r = 0; r < 8; r++) {
    int row = row0 + r + 8 * hi;      // row = b*S + s
    kh[(size_t)row * HP + n]      = (_Float16)ck0[r];
    kh[(size_t)row * HP + 16 + n] = (_Float16)ck1[r];
    int b = row >> 11, s = row & (Sn - 1);
    vt[((size_t)b * HP + n) * Sn + s]      = (_Float16)cv0[r];
    vt[((size_t)b * HP + 16 + n) * Sn + s] = (_Float16)cv1[r];
  }
}

// ------- Kernel 3: per-(b,s) softmax stats over t (base-2): m[b,s], invd[b,s] ---------
__global__ void k_stats(const _Float16* __restrict__ qh, const _Float16* __restrict__ kh,
                        float* __restrict__ mcol, float* __restrict__ invd) {
  int wid = blockIdx.x * (blockDim.x >> 5) + (threadIdx.x >> 5);
  int b = wid >> 7, s0 = (wid & 127) * 16;
  int L = threadIdx.x & 31, n = L & 15;
  const _Float16* qhb = qh + (size_t)b * Tn * HP;
  const _Float16* khb = kh + (size_t)b * Sn * HP;
  v16h bk = load_b16(khb, HP, s0, 0);   // loop-invariant B (k columns, K=32)
  float mrun = -1e30f, srun = 0.f;
  for (int t0 = 0; t0 < Tn; t0 += 16) {
    v8f c = {};
    c = wmma16(load_a16(qhb, HP, t0, 0), bk, c);
    float tmax = c[0];
#pragma unroll
    for (int r = 1; r < 8; r++) tmax = fmaxf(tmax, c[r]);
    float mnew = fmaxf(mrun, tmax);
    float ssum = 0.f;
#pragma unroll
    for (int r = 0; r < 8; r++) ssum += fast_exp2(c[r] - mnew);
    srun = srun * fast_exp2(mrun - mnew) + ssum;
    mrun = mnew;
  }
  float mo = __shfl_xor(mrun, 16, 32);
  float so = __shfl_xor(srun, 16, 32);
  float mt = fmaxf(mrun, mo);
  float st = srun * fast_exp2(mrun - mt) + so * fast_exp2(mo - mt);
  if (L < 16) {
    mcol[(size_t)b * Sn + s0 + n] = mt;
    invd[(size_t)b * Sn + s0 + n] = 1.f / st;
  }
}

// ------- Kernel 3b: fold 1/denom into V once: vt[b][h][s] *= invd[b][s] ---------------
__global__ void k_vscale(_Float16* __restrict__ vt, const float* __restrict__ invd) {
  int i = blockIdx.x * blockDim.x + threadIdx.x;  // 0 .. Bn*HP*Sn-1
  int s = i & (Sn - 1);
  int b = i >> 16;                                 // i >> (11+5)
  float sc = invd[(size_t)b * Sn + s];
  vt[i] = (_Float16)((float)vt[i] * sc);
}

// ------- Kernel 4: h1 = softmax(scores)·V', LDS-staged via async copies ---------------
// All 8 waves of a block share batch b and sweep identical kh/vt chunks; double-buffered
// global_load_async_to_lds_b128 staging, s_wait_asynccnt + barrier per chunk.
__global__ void __launch_bounds__(256) k_attn(const _Float16* __restrict__ qh,
                                              const _Float16* __restrict__ kh,
                                              const _Float16* __restrict__ vt,
                                              const float* __restrict__ mcol,
                                              _Float16* __restrict__ h1) {
  extern __shared__ _Float16 smem[];   // [0, SVTE): kh x2 bufs ; [SVTE, ...): vt x2 bufs
  int wid = blockIdx.x * 8 + (threadIdx.x >> 5);
  int b = wid >> 7, t0 = (wid & 127) * 16;
  int L = threadIdx.x & 31, n = L & 15, hi = L >> 4;
  const _Float16* qhb = qh + (size_t)b * Tn * HP;
  const _Float16* khb = kh + (size_t)b * Sn * HP;
  const _Float16* vtb = vt + (size_t)b * HP * Sn;
  const float* mb = mcol + (size_t)b * Sn;

  v16h bq = load_b16(qhb, HP, t0, 0);   // loop-invariant B (q columns, K=32)
  v8f c0 = {}, c1 = {};

  // stage chunk [s0, s0+CH) into buffer `buf` (4 async ops per thread)
  auto stage = [&](int s0, int buf) {
    int tid = threadIdx.x;
    const _Float16* gk = khb + (size_t)s0 * HP;
    const _Float16* gv = vtb + s0;
#pragma unroll
    for (int k = 0; k < 2; k++) {                    // kh: CH rows x 64B = 512 x16B
      int c = tid + (k << 8);
      int row = c >> 2, q = c & 3;
      unsigned l = (unsigned)((buf * CH * KROW + row * KROW + q * 8) * 2);
      async_cp16(l, gk + (size_t)row * HP + q * 8);
    }
#pragma unroll
    for (int k = 0; k < 2; k++) {                    // vt: 32 rows x 256B = 512 x16B
      int c = tid + (k << 8);
      int hr = c >> 4, q = c & 15;
      unsigned l = (unsigned)((SVTE + buf * HP * VROW + hr * VROW + q * 8) * 2);
      async_cp16(l, gv + (size_t)hr * Sn + q * 8);
    }
  };

  stage(0, 0);
  for (int ch = 0; ch < NCHUNK; ch++) {
    if (ch + 1 < NCHUNK) {
      stage((ch + 1) * CH, (ch + 1) & 1);
      asm volatile("s_wait_asynccnt 0x4" ::: "memory");  // current chunk landed
    } else {
      asm volatile("s_wait_asynccnt 0x0" ::: "memory");
    }
    __syncthreads();                                     // publish all waves' copies
    const _Float16* skh = smem + (size_t)(ch & 1) * CH * KROW;
    const _Float16* svt = smem + SVTE + (size_t)(ch & 1) * HP * VROW;
#pragma unroll 2
    for (int sl = 0; sl < CH; sl += 32) {
      v8f z1 = {}, z2 = {};
      v8f d1 = wmma16(load_a16(skh, KROW, sl, 0), bq, z1);       // scores (s x t)
      v8f d2 = wmma16(load_a16(skh, KROW, sl + 16, 0), bq, z2);
      int sb = ch * CH + sl + 8 * hi;
      v16h pa;
#pragma unroll
      for (int r = 0; r < 8; r++) {
        pa[r]     = (_Float16)fast_exp2(d1[r] - mb[sb + r]);       // K = sl + 8*hi + r
        pa[r + 8] = (_Float16)fast_exp2(d2[r] - mb[sb + 16 + r]);  // K = sl+16+8*hi+r
      }
      v16h bv0 = load16h_2(svt + (size_t)n * VROW + sl + 8 * hi);
      v16h bv1 = load16h_2(svt + (size_t)(n + 16) * VROW + sl + 8 * hi);
      c0 = wmma16(pa, bv0, c0);
      c1 = wmma16(pa, bv1, c1);
    }
    __syncthreads();                                     // reads done before overwrite
  }
#pragma unroll
  for (int r = 0; r < 8; r++) {
    int t = t0 + r + 8 * hi;
    h1[((size_t)b * Tn + t) * HP + n]      = (_Float16)c0[r];
    h1[((size_t)b * Tn + t) * HP + 16 + n] = (_Float16)c1[r];
  }
}

// ------- Kernel 5: out = h1 @ wh (K=32 single chunk, N=1024) --------------------------
__global__ void k_out(const _Float16* __restrict__ h1, const _Float16* __restrict__ wht,
                      float* __restrict__ out) {
  int wid = blockIdx.x * (blockDim.x >> 5) + (threadIdx.x >> 5);
  int mt = wid >> 4, nc = wid & 15;
  int row0 = mt * 16;
  int L = threadIdx.x & 31, n = L & 15, hi = L >> 4;
  v16h a = load_a16(h1, HP, row0, 0);
#pragma unroll
  for (int j = 0; j < 4; j++) {
    int n0 = nc * 64 + j * 16;
    v8f c = {};
    c = wmma16(a, load_b16(wht, HP, n0, 0), c);
#pragma unroll
    for (int r = 0; r < 8; r++)
      out[(size_t)(row0 + r + 8 * hi) * Dn + n0 + n] = c[r];
  }
}

extern "C" void kernel_launch(void* const* d_in, const int* in_sizes, int n_in,
                              void* d_out, int out_size, void* d_ws, size_t ws_size,
                              hipStream_t stream) {
  const float* encoded = (const float*)d_in[0];
  const float* text    = (const float*)d_in[1];
  const float* wq      = (const float*)d_in[2];
  const float* wk      = (const float*)d_in[3];
  const float* wv      = (const float*)d_in[4];
  const float* wh      = (const float*)d_in[5];
  float* out = (float*)d_out;

  char* ws = (char*)d_ws;
  size_t o = 0;
  _Float16* qh  = (_Float16*)(ws + o); o += (size_t)Bn * Tn * HP * 2;  // 2 MB
  _Float16* kh  = (_Float16*)(ws + o); o += (size_t)Bn * Sn * HP * 2;  // 2 MB
  _Float16* vt  = (_Float16*)(ws + o); o += (size_t)Bn * HP * Sn * 2;  // 2 MB
  _Float16* h1  = (_Float16*)(ws + o); o += (size_t)Bn * Tn * HP * 2;  // 2 MB
  _Float16* wqt = (_Float16*)(ws + o); o += (size_t)HP * Dn * 2;       // 64 KB
  _Float16* wkt = (_Float16*)(ws + o); o += (size_t)HP * Dn * 2;
  _Float16* wvt = (_Float16*)(ws + o); o += (size_t)HP * Dn * 2;
  _Float16* wht = (_Float16*)(ws + o); o += (size_t)Dn * HP * 2;
  float* mcol   = (float*)(ws + o);    o += (size_t)Bn * Sn * 4;       // 128 KB
  float* invd   = (float*)(ws + o);    o += (size_t)Bn * Sn * 4;

  k_wconv<<<(HP * Dn) / 256, 256, 0, stream>>>(wq, wk, wv, wh, wqt, wkt, wvt, wht);
  k_projq<<<256, 256, 0, stream>>>(text, wqt, qh);
  k_projkv<<<256, 256, 0, stream>>>(encoded, wkt, wvt, kh, vt);
  k_stats<<<256, 256, 0, stream>>>(qh, kh, mcol, invd);
  k_vscale<<<(Bn * HP * Sn) / 256, 256, 0, stream>>>(vt, invd);
  k_attn<<<256, 256, LDS_BYTES, stream>>>(qh, kh, vt, mcol, h1);
  k_out<<<4096, 256, 0, stream>>>(h1, wht, out);
}
